// MultiHeadAttention_58841051955535
// MI455X (gfx1250) — matrix operand order repaired and checked by reading, not compiled, not run on hardware
//
#include <hip/hip_runtime.h>

// MHA for MI455X (gfx1250): B=4, S=1024, D=1024, H=16, DK=DV=64.
// GEMM stages on v_wmma_f32_16x16x32_bf16 (f32 accum); softmax in f32.
// Q/K tiles in the attention kernel are DMA'd into LDS by the Tensor Data
// Mover (tensor_load_to_lds + s_wait_tensorcnt). Fragments are 2x b128 LDS
// loads thanks to contiguous per-lane K runs and 8-element-aligned strides.

#define B_  4
#define S_  1024
#define D_  1024
#define H_  16
#define DK_ 64
#define DV_ 64

typedef __attribute__((ext_vector_type(16))) __bf16        v16bf;
typedef __attribute__((ext_vector_type(8)))  float         v8f;
typedef __attribute__((ext_vector_type(4)))  unsigned int  su4;
typedef __attribute__((ext_vector_type(8)))  unsigned int  su8;

union Frag  { v16bf v; su4 q[2]; };
union Pack8 { su4 q; __bf16 h[8]; };
union Pack4 { unsigned int u[2]; __bf16 h[4]; };

__device__ __forceinline__ v8f wmma_bf16(v16bf a, v16bf b, v8f c) {
  return __builtin_amdgcn_wmma_f32_16x16x32_bf16(false, a, false, b, (short)0, c,
                                                 false, false);
}

// A fragment (16x32 bf16), element (m,k) from row-major lds[m*stride + k].
// Per-lane: k = hi*16 + i  -> two b128 loads.
// Requires stride % 8 == 0 and kbase % 8 == 0.
__device__ __forceinline__ v16bf frag_a(const __bf16* p, int mbase, int stride,
                                        int kbase, int lane) {
  const int hi = (lane >> 4) & 1;
  const __bf16* r = p + (mbase + (lane & 15)) * stride + kbase + hi * 16;
  Frag f;
  f.q[0] = *(const su4*)(r);
  f.q[1] = *(const su4*)(r + 8);
  return f.v;
}

// B fragment (32x16), element (k,n) with the matrix staged transposed in LDS
// as [n][k] row-major. Per-lane: k = kbase + hi*16 + i  -> two b128 loads.
__device__ __forceinline__ v16bf frag_b(const __bf16* p, int nbase, int stride,
                                        int kbase, int lane) {
  const __bf16* r = p + (nbase + (lane & 15)) * stride + kbase + ((lane >> 4) & 1) * 16;
  Frag f;
  f.q[0] = *(const su4*)(r);
  f.q[1] = *(const su4*)(r + 8);
  return f.v;
}

// ---- Tensor Data Mover: 2D 64x64 bf16 tile, row stride 64, LDS-padded to 72.
// D# group0: count=1, lds_addr, global_addr, type=2.
// D# group1: data_size=2B, pad_enable, pad_interval=128B (code 4),
//            pad_amount=4 dwords (code 3); dims/tile = 64; dim0 stride = 64.
__device__ __forceinline__ void tdm_load_tile64_bf16(unsigned lds_byte_addr,
                                                     const void* gaddr) {
  unsigned long long ga = (unsigned long long)(size_t)gaddr;
  su4 g0;
  g0[0] = 1u;
  g0[1] = lds_byte_addr;
  g0[2] = (unsigned)ga;
  g0[3] = ((unsigned)(ga >> 32) & 0x01FFFFFFu) | 0x80000000u;
  su8 g1;
  g1[0] = (1u << 16) | (1u << 20) | (4u << 22) | (3u << 25);
  g1[1] = 64u << 16;   // tensor_dim0 = 64 (bits 63:48)
  g1[2] = 64u << 16;   // tensor_dim1 = 64 (bits 95:80)
  g1[3] = 64u << 16;   // tile_dim0   = 64 (bits 127:112)
  g1[4] = 64u;         // tile_dim1   = 64 (bits 143:128)
  g1[5] = 64u;         // tensor_dim0_stride = 64 (bits 191:160)
  g1[6] = 0u;
  g1[7] = 0u;
  asm volatile("tensor_load_to_lds %0, %1" :: "s"(g0), "s"(g1) : "memory");
}

__device__ __forceinline__ unsigned lds_offset(const void* p) {
  // Generic LDS address: addr[31:0] is the in-LDS byte offset.
  return (unsigned)(size_t)p;
}

// ---------------------------------------------------------------------------
// Per-head projection: Y[b,h,s,0:64] = X[b,s,:] @ W[h]  (f32 in, bf16 out)
// ---------------------------------------------------------------------------
__global__ void __launch_bounds__(256)
mha_proj_kernel(const float* __restrict__ X,   // [B*S, D]
                const float* __restrict__ W,   // [H, D, 64]
                __bf16* __restrict__ Y) {      // [B, H, S, 64]
  const int mtile = blockIdx.x;                // 64 rows each
  const int h = blockIdx.y;
  const int tid = threadIdx.x, lane = tid & 31, wv = tid >> 5;
  const int mt = wv >> 1, nb = (wv & 1) * 32;

  __shared__ alignas(16) __bf16 lA[64 * 40];   // [m][k]
  __shared__ alignas(16) __bf16 lBt[64 * 40];  // [n][k] (transposed W chunk)

  const int m0 = mtile * 64;
  const float* Wh = W + (size_t)h * (D_ * DK_);
  v8f c0 = {}; v8f c1 = {};

  for (int kk = 0; kk < D_; kk += 32) {
    __syncthreads();
#pragma unroll
    for (int t = 0; t < 2; ++t) {
      int idx = t * 256 + tid;                 // 0..511
      {  // A: 64x32 f32 -> bf16, row-major
        int r = idx >> 3, c4 = (idx & 7) * 4;
        float4 xv = *(const float4*)(X + (size_t)(m0 + r) * D_ + kk + c4);
        Pack4 pk;
        pk.h[0] = (__bf16)xv.x; pk.h[1] = (__bf16)xv.y;
        pk.h[2] = (__bf16)xv.z; pk.h[3] = (__bf16)xv.w;
        *(unsigned int*)&lA[r * 40 + c4]     = pk.u[0];
        *(unsigned int*)&lA[r * 40 + c4 + 2] = pk.u[1];
      }
      {  // B: 32x64 f32 -> bf16, transposed into [n][k]
        int r = idx >> 4, c4 = (idx & 15) * 4;
        float4 wv4 = *(const float4*)(Wh + (size_t)(kk + r) * DK_ + c4);
        lBt[(c4 + 0) * 40 + r] = (__bf16)wv4.x;
        lBt[(c4 + 1) * 40 + r] = (__bf16)wv4.y;
        lBt[(c4 + 2) * 40 + r] = (__bf16)wv4.z;
        lBt[(c4 + 3) * 40 + r] = (__bf16)wv4.w;
      }
    }
    __syncthreads();
    v16bf a  = frag_a(lA, mt * 16, 40, 0, lane);
    v16bf b0 = frag_b(lBt, nb, 40, 0, lane);
    v16bf b1 = frag_b(lBt, nb + 16, 40, 0, lane);
    c0 = wmma_bf16(a, b0, c0);
    c1 = wmma_bf16(a, b1, c1);
  }

  const int bb = mtile >> 4, s0 = (mtile & 15) << 6;
  __bf16* Yb = Y + (((size_t)bb * H_ + h) * S_ + s0) * DK_;
  const int hi = (lane >> 4) & 1, nl = lane & 15;
#pragma unroll
  for (int r = 0; r < 8; ++r) {
    int m = mt * 16 + r + hi * 8;
    Yb[(size_t)m * DK_ + nb + nl]      = (__bf16)c0[r];
    Yb[(size_t)m * DK_ + nb + 16 + nl] = (__bf16)c1[r];
  }
}

// ---------------------------------------------------------------------------
// Flash attention over one (b, h, 64-query tile). Q/K tiles arrive via TDM.
// Masked value is -1e10 so fully-dead rows reproduce the reference's uniform
// softmax. Output is head-major bf16: O[b, s, h*64 + dv].
// ---------------------------------------------------------------------------
__global__ void __launch_bounds__(256)
mha_attn_kernel(const __bf16* __restrict__ Q, const __bf16* __restrict__ K,
                const __bf16* __restrict__ V, const int* __restrict__ pad1v,
                const int* __restrict__ pad2v, __bf16* __restrict__ O) {
  const int qt = blockIdx.x, h = blockIdx.y, b = blockIdx.z;
  const int tid = threadIdx.x, lane = tid & 31, wv = tid >> 5;
  const int mt = wv >> 1, nb = (wv & 1) * 32;
  const int hi = (lane >> 4) & 1, nl = lane & 15;
  const int q0 = qt * 64;

  __shared__ alignas(16) __bf16 Qt[64 * 72];   // [query][dk]   (TDM, padded)
  __shared__ alignas(16) __bf16 Kt[64 * 72];   // [key][dk]     (TDM, padded)
  __shared__ alignas(16) __bf16 VtT[64 * 72];  // [dv][key]     (manual transpose)
  __shared__ alignas(16) __bf16 Pt[64 * 72];   // [query][key]
  __shared__ alignas(16) float  Sc[64 * 68];
  __shared__ float rowm[64], rowl[64], rowa[64];

  const size_t headBase = ((size_t)b * H_ + h) * (size_t)S_ * DK_;
  const __bf16* Qh = Q + headBase + (size_t)q0 * DK_;

  if (wv == 0) tdm_load_tile64_bf16(lds_offset(Qt), Qh);   // async Q tile
  if (tid < 64) { rowm[tid] = -3.0e38f; rowl[tid] = 0.0f; }
  const int pad1 = pad1v[b], pad2 = pad2v[b];
  v8f o0 = {}; v8f o1 = {};

  for (int kb = 0; kb < S_; kb += 64) {
    if (wv == 0) tdm_load_tile64_bf16(lds_offset(Kt), K + headBase + (size_t)kb * DK_);
    // V tile, transposed into [dv][key]
#pragma unroll
    for (int t = 0; t < 2; ++t) {
      int idx = t * 256 + tid;
      int r = idx >> 3, c8 = (idx & 7) * 8;
      Pack8 vv;
      vv.q = *(const su4*)(V + headBase + (size_t)(kb + r) * DK_ + c8);
#pragma unroll
      for (int j = 0; j < 8; ++j) VtT[(c8 + j) * 72 + r] = vv.h[j];
    }
    if (wv == 0) __builtin_amdgcn_s_wait_tensorcnt(0);
    __syncthreads();

    // scores = Q @ K^T over dk = 64 (two K-chunks of 32)
    v8f sA = {}; v8f sB = {};
#pragma unroll
    for (int kc = 0; kc < 64; kc += 32) {
      v16bf a  = frag_a(Qt, mt * 16, 72, kc, lane);
      v16bf b0 = frag_b(Kt, nb, 72, kc, lane);
      v16bf b1 = frag_b(Kt, nb + 16, 72, kc, lane);
      sA = wmma_bf16(a, b0, sA);
      sB = wmma_bf16(a, b1, sB);
    }

    // scale + masks -> Sc (f32)
#pragma unroll
    for (int r = 0; r < 8; ++r) {
      int m = mt * 16 + r + hi * 8;
      int qrow = q0 + m;
      int j0 = kb + nb + nl, j1 = j0 + 16;
      float s0v = sA[r] * 0.125f;   // 1/sqrt(64)
      float s1v = sB[r] * 0.125f;
      bool rowdead = (qrow >= pad2);
      if (rowdead || j0 >= pad1 || j0 > qrow) s0v = -1e10f;
      if (rowdead || j1 >= pad1 || j1 > qrow) s1v = -1e10f;
      Sc[m * 68 + nb + nl]      = s0v;
      Sc[m * 68 + nb + 16 + nl] = s1v;
    }
    __syncthreads();

    // online softmax: one thread per row
    if (tid < 64) {
      int m = tid;
      float mold = rowm[m];
      float bm = -3.0e38f;
      for (int j = 0; j < 64; ++j) bm = fmaxf(bm, Sc[m * 68 + j]);
      float mnew = fmaxf(mold, bm);
      float alpha = __expf(mold - mnew);
      float lsum = 0.0f;
      for (int j = 0; j < 64; ++j) {
        float pv = __expf(Sc[m * 68 + j] - mnew);
        Pt[m * 72 + j] = (__bf16)pv;
        lsum += pv;
      }
      rowl[m] = rowl[m] * alpha + lsum;
      rowm[m] = mnew;
      rowa[m] = alpha;
    }
    __syncthreads();

    // O = O*alpha + P @ V
#pragma unroll
    for (int r = 0; r < 8; ++r) {
      float al = rowa[mt * 16 + r + hi * 8];
      o0[r] *= al;
      o1[r] *= al;
    }
#pragma unroll
    for (int kc = 0; kc < 64; kc += 32) {
      v16bf a  = frag_a(Pt, mt * 16, 72, kc, lane);
      v16bf b0 = frag_b(VtT, nb, 72, kc, lane);
      v16bf b1 = frag_b(VtT, nb + 16, 72, kc, lane);
      o0 = wmma_bf16(a, b0, o0);
      o1 = wmma_bf16(a, b1, o1);
    }
    __syncthreads();
  }

#pragma unroll
  for (int r = 0; r < 8; ++r) {
    int m = mt * 16 + r + hi * 8;
    float inv = 1.0f / rowl[m];
    size_t base = ((size_t)b * S_ + (q0 + m)) * (size_t)(H_ * DV_) + (size_t)h * DV_;
    O[base + nb + nl]      = (__bf16)(o0[r] * inv);
    O[base + nb + 16 + nl] = (__bf16)(o1[r] * inv);
  }
}

// ---------------------------------------------------------------------------
// Output projection: out[4096,1024] = Obf[4096,1024] @ Wo[1024,1024] (f32 out)
// ---------------------------------------------------------------------------
__global__ void __launch_bounds__(256)
mha_outproj_kernel(const __bf16* __restrict__ A, const float* __restrict__ Wo,
                   float* __restrict__ out) {
  const int mtile = blockIdx.x, ntile = blockIdx.y;
  const int tid = threadIdx.x, lane = tid & 31, wv = tid >> 5;
  const int mt = wv >> 1, nb = (wv & 1) * 32;
  const int m0 = mtile * 64, n0 = ntile * 64;

  __shared__ alignas(16) __bf16 lA[64 * 40];   // [m][k]
  __shared__ alignas(16) __bf16 lBt[64 * 40];  // [n][k]

  v8f c0 = {}; v8f c1 = {};
  for (int kk = 0; kk < D_; kk += 32) {
    __syncthreads();
    {  // A: 64x32 bf16, one b128 per thread
      int r = tid >> 2, c8 = (tid & 3) * 8;
      *(su4*)&lA[r * 40 + c8] =
          *(const su4*)(A + (size_t)(m0 + r) * D_ + kk + c8);
    }
#pragma unroll
    for (int t = 0; t < 2; ++t) {  // Wo: 32x64 f32 -> bf16 transposed
      int idx = t * 256 + tid;
      int r = idx >> 4, c4 = (idx & 15) * 4;
      float4 wv4 = *(const float4*)(Wo + (size_t)(kk + r) * D_ + n0 + c4);
      lBt[(c4 + 0) * 40 + r] = (__bf16)wv4.x;
      lBt[(c4 + 1) * 40 + r] = (__bf16)wv4.y;
      lBt[(c4 + 2) * 40 + r] = (__bf16)wv4.z;
      lBt[(c4 + 3) * 40 + r] = (__bf16)wv4.w;
    }
    __syncthreads();
    v16bf a  = frag_a(lA, mt * 16, 40, 0, lane);
    v16bf b0 = frag_b(lBt, nb, 40, 0, lane);
    v16bf b1 = frag_b(lBt, nb + 16, 40, 0, lane);
    c0 = wmma_bf16(a, b0, c0);
    c1 = wmma_bf16(a, b1, c1);
  }

  const int hi = (lane >> 4) & 1, nl = lane & 15;
#pragma unroll
  for (int r = 0; r < 8; ++r) {
    int m = m0 + mt * 16 + r + hi * 8;
    out[(size_t)m * D_ + n0 + nb + nl]      = c0[r];
    out[(size_t)m * D_ + n0 + nb + 16 + nl] = c1[r];
  }
}

extern "C" void kernel_launch(void* const* d_in, const int* in_sizes, int n_in,
                              void* d_out, int out_size, void* d_ws, size_t ws_size,
                              hipStream_t stream) {
  const float* q  = (const float*)d_in[0];
  const float* k  = (const float*)d_in[1];
  const float* v  = (const float*)d_in[2];
  const int*   p1 = (const int*)d_in[3];
  const int*   p2 = (const int*)d_in[4];
  const float* Wq = (const float*)d_in[5];
  const float* Wk = (const float*)d_in[6];
  const float* Wv = (const float*)d_in[7];
  const float* Wo = (const float*)d_in[8];
  float* out = (float*)d_out;

  const size_t headElems = (size_t)B_ * H_ * S_ * DK_;  // 4M bf16 per tensor
  __bf16* qh = (__bf16*)d_ws;
  __bf16* kh = qh + headElems;
  __bf16* vh = kh + headElems;
  __bf16* oh = vh + headElems;   // [B, S, H*DV] bf16

  dim3 blk(256);
  dim3 gp(64, H_);
  mha_proj_kernel<<<gp, blk, 0, stream>>>(q, Wq, qh);
  mha_proj_kernel<<<gp, blk, 0, stream>>>(k, Wk, kh);
  mha_proj_kernel<<<gp, blk, 0, stream>>>(v, Wv, vh);

  dim3 ga(S_ / 64, H_, B_);
  mha_attn_kernel<<<ga, blk, 0, stream>>>(qh, kh, vh, p1, p2, oh);

  dim3 go(64, D_ / 64);
  mha_outproj_kernel<<<go, blk, 0, stream>>>(oh, Wo, out);

  (void)in_sizes; (void)n_in; (void)out_size; (void)ws_size;
}
